// HeteroRGCN_35682588295609
// MI455X (gfx1250) — compile-verified, matrix operand order checked by reading, not run on hardware
//
#include <hip/hip_runtime.h>
#include <hip/hip_bf16.h>

typedef __attribute__((ext_vector_type(16))) _Float16 v16h;
typedef __attribute__((ext_vector_type(8)))  float    v8f;

#define D_FEAT 128
#define LDSP   136   // padded LDS row stride (halves); 272B row stride stays 16B-aligned

// ---------------- GEMM + bias: C[M x 128] = A[M x 128] @ W[128 x 128] + b ----------------
// block = 256 threads = 8 wave32; block covers 64 rows; wave w owns N-tile [16w,16w+16)
// W is staged TRANSPOSED in LDS so B fragments load as contiguous b128, like A.
__global__ __launch_bounds__(256) void rgcn_gemm_bias(
    const float* __restrict__ A, const float* __restrict__ W,
    const float* __restrict__ bias, float* __restrict__ C, int M)
{
    __shared__ _Float16 Al[64 * LDSP];     // Al[row][k]
    __shared__ _Float16 Wt[128 * LDSP];    // Wt[col][k]  (transposed W)
    const int tid  = threadIdx.x;
    const int row0 = blockIdx.x * 64;

    // stage A strip (f32 -> f16), row-major
    for (int idx = tid; idx < 64 * D_FEAT; idx += 256) {
        int r = idx >> 7, c = idx & 127;
        int gr = row0 + r;
        float v = (gr < M) ? A[(size_t)gr * D_FEAT + c] : 0.0f;
        Al[r * LDSP + c] = (_Float16)v;
    }
    // stage W (f32 -> f16) with transpose: coalesced global reads, scattered LDS b16 writes
    for (int idx = tid; idx < D_FEAT * D_FEAT; idx += 256) {
        int r = idx >> 7, c = idx & 127;   // W[r][c]
        Wt[c * LDSP + r] = (_Float16)W[idx];
    }
    __syncthreads();

    const int wave = tid >> 5;
    const int lane = tid & 31;
    const int half = lane >> 4;   // 0: lanes 0-15, 1: lanes 16-31
    const int lr   = lane & 15;
    const int col  = wave * 16 + lr;
    const float bv = bias[col];
    const bool full = (row0 + 64 <= M);

    // hoist the 4 strip-invariant B fragments (K = 0,32,64,96)
    v16h bf[4];
    #pragma unroll
    for (int kk = 0; kk < 4; ++kk) {
        const _Float16* bp = &Wt[col * LDSP + kk * 32 + half * 8];
        #pragma unroll
        for (int j = 0; j < 8; ++j) { bf[kk][j] = bp[j]; bf[kk][j + 8] = bp[j + 16]; }
    }

    for (int strip = 0; strip < 4; ++strip) {
        v8f c;
        #pragma unroll
        for (int i = 0; i < 8; ++i) c[i] = bv;   // bias broadcast along M

        #pragma unroll
        for (int kk = 0; kk < 4; ++kk) {
            v16h a;
            // A fragment: lane holds row (lr), K = 32kk + half*8 .. +7 and +16..+23
            const _Float16* ap = &Al[(strip * 16 + lr) * LDSP + kk * 32 + half * 8];
            #pragma unroll
            for (int j = 0; j < 8; ++j) { a[j] = ap[j]; a[j + 8] = ap[j + 16]; }

            c = __builtin_amdgcn_wmma_f32_16x16x32_f16(
                    false, a, false, bf[kk], (short)0, c, false, false);
        }
        // C/D layout: VGPR i -> M = i + half*8, N = lr (within tile)
        float* cp = C + (size_t)(row0 + strip * 16 + half * 8) * D_FEAT + col;
        if (full) {
            #pragma unroll
            for (int i = 0; i < 8; ++i) cp[(size_t)i * D_FEAT] = c[i];
        } else {
            #pragma unroll
            for (int i = 0; i < 8; ++i) {
                int gm = row0 + strip * 16 + half * 8 + i;
                if (gm < M) cp[(size_t)i * D_FEAT] = c[i];
            }
        }
    }
}

// ---------------- per-dst in-degree count, then reciprocal ----------------
__global__ void count_edges(const int* __restrict__ dst, unsigned* __restrict__ cnt, int E) {
    int i = blockIdx.x * blockDim.x + threadIdx.x;
    if (i < E) atomicAdd(&cnt[dst[i]], 1u);
}
__global__ void inv_counts(float* buf, int n) {   // in-place u32 -> 1/max(c,1)
    int i = blockIdx.x * blockDim.x + threadIdx.x;
    if (i < n) {
        unsigned c = ((unsigned*)buf)[i];
        buf[i] = 1.0f / (float)(c ? c : 1u);
    }
}

// ---------------- mean-scaled scatter: out[dst] += Wh[src] * inv[dst] ----------------
// one wave32 per edge, float4 per lane (coalesced 512B)
__global__ __launch_bounds__(256) void scatter_mean(
    const float* __restrict__ Wh, const int* __restrict__ src,
    const int* __restrict__ dst, const float* __restrict__ inv,
    float* __restrict__ out, int E)
{
    int lane = threadIdx.x & 31;
    int e = blockIdx.x * 8 + (threadIdx.x >> 5);
    if (e >= E) return;
    int s = src[e], d = dst[e];
    float sc = inv[d];
    float4 m = ((const float4*)(Wh + (size_t)s * D_FEAT))[lane];
    float* po = out + (size_t)d * D_FEAT + lane * 4;
    unsafeAtomicAdd(po + 0, m.x * sc);
    unsafeAtomicAdd(po + 1, m.y * sc);
    unsafeAtomicAdd(po + 2, m.z * sc);
    unsafeAtomicAdd(po + 3, m.w * sc);
}

__global__ void leaky_relu_k(float* x, int n) {
    int i = blockIdx.x * blockDim.x + threadIdx.x;
    if (i < n) { float v = x[i]; x[i] = v > 0.0f ? v : 0.01f * v; }
}

// ---------------- fused pos+neg dot-product scores ----------------
// one wave32 per edge; reuses the h[s][src] gather for both pos and neg
__global__ __launch_bounds__(256) void dot_sig2(
    const float* __restrict__ hs, const float* __restrict__ hd,
    const int* __restrict__ src, const int* __restrict__ dst,
    const int* __restrict__ ndst, float* __restrict__ outp,
    float* __restrict__ outn, int E)
{
    int lane = threadIdx.x & 31;
    int e = blockIdx.x * 8 + (threadIdx.x >> 5);
    if (e >= E) return;
    int s = src[e], d = dst[e], nd = ndst[e];
    float4 u = ((const float4*)(hs + (size_t)s  * D_FEAT))[lane];
    float4 v = ((const float4*)(hd + (size_t)d  * D_FEAT))[lane];
    float4 w = ((const float4*)(hd + (size_t)nd * D_FEAT))[lane];
    float sp = u.x * v.x + u.y * v.y + u.z * v.z + u.w * v.w;
    float sn = u.x * w.x + u.y * w.y + u.z * w.z + u.w * w.w;
    #pragma unroll
    for (int m = 16; m >= 1; m >>= 1) {
        sp += __shfl_xor(sp, m, 32);
        sn += __shfl_xor(sn, m, 32);
    }
    if (lane == 0) {
        outp[e] = 1.0f / (1.0f + __expf(-sp));
        outn[e] = 1.0f / (1.0f + __expf(-sn));
    }
}

extern "C" void kernel_launch(void* const* d_in, const int* in_sizes, int n_in,
                              void* d_out, int out_size, void* d_ws, size_t ws_size,
                              hipStream_t stream) {
    // ---- inputs (setup_inputs order) ----
    const float* feat_drug = (const float*)d_in[0];
    const float* feat_gene = (const float*)d_in[1];
    const float* feat_dis  = (const float*)d_in[2];
    const float* W1[4] = {(const float*)d_in[3], (const float*)d_in[5],
                          (const float*)d_in[7], (const float*)d_in[9]};
    const float* b1[4] = {(const float*)d_in[4], (const float*)d_in[6],
                          (const float*)d_in[8], (const float*)d_in[10]};
    const float* W2[4] = {(const float*)d_in[11], (const float*)d_in[13],
                          (const float*)d_in[15], (const float*)d_in[17]};
    const float* b2[4] = {(const float*)d_in[12], (const float*)d_in[14],
                          (const float*)d_in[16], (const float*)d_in[18]};
    // etype order: treats(drug->disease), targets(drug->gene),
    //              assoc(gene->disease), rev_treats(disease->drug)
    const int* src_e[4]  = {(const int*)d_in[19], (const int*)d_in[22],
                            (const int*)d_in[25], (const int*)d_in[28]};
    const int* dst_e[4]  = {(const int*)d_in[20], (const int*)d_in[23],
                            (const int*)d_in[26], (const int*)d_in[29]};
    const int* ndst_e[4] = {(const int*)d_in[21], (const int*)d_in[24],
                            (const int*)d_in[27], (const int*)d_in[30]};

    const int ND  = in_sizes[0] / D_FEAT;   // 20000 drug
    const int NG  = in_sizes[1] / D_FEAT;   // 80000 gene
    const int NDI = in_sizes[2] / D_FEAT;   // 20000 disease
    const int E   = in_sizes[19];           // 800000
    float* out = (float*)d_out;

    // ---- workspace carve-up ----
    float* ws = (float*)d_ws;
    size_t o = 0;
    float* h1_drug = ws + o; o += (size_t)ND  * D_FEAT;
    float* h1_gene = ws + o; o += (size_t)NG  * D_FEAT;
    float* h1_dis  = ws + o; o += (size_t)NDI * D_FEAT;
    float* h2_drug = ws + o; o += (size_t)ND  * D_FEAT;
    float* h2_gene = ws + o; o += (size_t)NG  * D_FEAT;
    float* h2_dis  = ws + o; o += (size_t)NDI * D_FEAT;
    float* Wh      = ws + o; o += (size_t)NG  * D_FEAT;   // max N_s rows
    float* inv_tr  = ws + o; o += NDI;                    // treats: dst=disease
    float* inv_tg  = ws + o; o += NG;                     // targets: dst=gene
    float* inv_as  = ws + o; o += NDI;                    // assoc: dst=disease
    float* inv_rv  = ws + o; o += ND;                     // rev_treats: dst=drug
    float* invp[4] = {inv_tr, inv_tg, inv_as, inv_rv};
    int    invn[4] = {NDI, NG, NDI, ND};

    const size_t h_tot = (size_t)(ND + NG + NDI) * D_FEAT;
    const int edge_blocks = (E + 7) / 8;        // 1 wave per edge
    const int thr_blocks  = (E + 255) / 256;

    // ---- per-etype mean reciprocals (graph-structure only; shared by both layers) ----
    hipMemsetAsync(inv_tr, 0, (size_t)(2 * NDI + NG + ND) * sizeof(float), stream);
    for (int i = 0; i < 4; ++i)
        count_edges<<<thr_blocks, 256, 0, stream>>>(dst_e[i], (unsigned*)invp[i], E);
    for (int i = 0; i < 4; ++i)
        inv_counts<<<(invn[i] + 255) / 256, 256, 0, stream>>>(invp[i], invn[i]);

    // ---- layer 1 ----
    hipMemsetAsync(h1_drug, 0, h_tot * sizeof(float), stream);
    const float* feats1[4] = {feat_drug, feat_drug, feat_gene, feat_dis};
    int          nsrc1[4]  = {ND, ND, NG, NDI};
    float*       h1dst[4]  = {h1_dis, h1_gene, h1_dis, h1_drug};
    for (int i = 0; i < 4; ++i) {
        int M = nsrc1[i];
        rgcn_gemm_bias<<<(M + 63) / 64, 256, 0, stream>>>(feats1[i], W1[i], b1[i], Wh, M);
        scatter_mean<<<edge_blocks, 256, 0, stream>>>(Wh, src_e[i], dst_e[i], invp[i], h1dst[i], E);
    }
    leaky_relu_k<<<((int)h_tot + 255) / 256, 256, 0, stream>>>(h1_drug, (int)h_tot);

    // ---- layer 2 ----
    hipMemsetAsync(h2_drug, 0, h_tot * sizeof(float), stream);
    const float* feats2[4] = {h1_drug, h1_drug, h1_gene, h1_dis};
    float*       h2dst[4]  = {h2_dis, h2_gene, h2_dis, h2_drug};
    for (int i = 0; i < 4; ++i) {
        int M = nsrc1[i];
        rgcn_gemm_bias<<<(M + 63) / 64, 256, 0, stream>>>(feats2[i], W2[i], b2[i], Wh, M);
        scatter_mean<<<edge_blocks, 256, 0, stream>>>(Wh, src_e[i], dst_e[i], invp[i], h2dst[i], E);
    }

    // ---- fused pos+neg dot scores ----
    const float* hs[4] = {h2_drug, h2_drug, h2_gene, h2_dis};
    const float* hd[4] = {h2_dis, h2_gene, h2_dis, h2_drug};
    for (int i = 0; i < 4; ++i) {
        dot_sig2<<<edge_blocks, 256, 0, stream>>>(
            hs[i], hd[i], src_e[i], dst_e[i], ndst_e[i],
            out + (size_t)i * E, out + (size_t)(4 + i) * E, E);
    }
}
